// QConv2d_73547019977147
// MI455X (gfx1250) — matrix operand order repaired and checked by reading, not compiled, not run on hardware
//
#include <hip/hip_runtime.h>

// ---------------------------------------------------------------------------
// QConv2d on MI455X (gfx1250): implicit-GEMM conv via v_wmma_f32_16x16x32_bf16
//   out[n,o,h,w] = scale * sum_{c,kh,kw} x[n,c,h+kh-1,w+kw-1] * W[o,c,kh,kw] + bias[o]
// K = 32ch * 9taps -> 9 WMMA K-steps of 32. Activations staged as NHWC bf16
// in d_ws (64 MB, L2-resident on the 192MB-L2 MI455X). Weight B-tiles staged
// into LDS by the Tensor Data Mover (tensor_load_to_lds, TENSORcnt).
// ---------------------------------------------------------------------------

typedef __attribute__((ext_vector_type(16))) __bf16        v16bf;
typedef __attribute__((ext_vector_type(8)))  float         v8f;
typedef __attribute__((ext_vector_type(4)))  float         v4f;
typedef __attribute__((ext_vector_type(4)))  unsigned int  v4u;
typedef __attribute__((ext_vector_type(8)))  unsigned int  v8u;
typedef __attribute__((ext_vector_type(4)))  int           v4i;
typedef __attribute__((ext_vector_type(8)))  int           v8i;

#define XOFF_BYTES 32768      // NHWC bf16 activation region offset inside d_ws
#define C_IN  32
#define O_CH  32
#define HSZ   256
#define WSZ   256
#define WPK_DWORDS 4608       // 9 taps * 2 ntiles * 32 lanes * 8 dwords

__device__ __forceinline__ unsigned short f2bf(float f) {
  unsigned u = __builtin_bit_cast(unsigned, f);
  unsigned r = (u + 0x7FFFu + ((u >> 16) & 1u)) >> 16;   // round-to-nearest-even
  return (unsigned short)r;
}

// ---------------------------------------------------------------------------
// Kernel 1: pack OIHW fp32 weights into WMMA B-operand bf16 tiles.
// dword index = ((tap*2 + ntile)*32 + lane)*8 + i
//   B tile is K(32) x N(16):  N = (lane&15) + 16*ntile
//   lane half h: VGPR i<4 holds K pair 8h+2i, VGPR i>=4 holds K pair 16+8h+2(i-4)
// ---------------------------------------------------------------------------
__global__ __launch_bounds__(256) void pack_weights(const float* __restrict__ w,
                                                    unsigned int* __restrict__ wpk) {
  int idx  = blockIdx.x * 256 + threadIdx.x;   // 0..4607
  int i    = idx & 7;
  int lane = (idx >> 3) & 31;
  int jt   = idx >> 8;                         // tap*2 + ntile, 0..17
  int j    = jt & 1;
  int tap  = jt >> 1;
  int n    = (lane & 15) + 16 * j;             // output channel
  int c0   = ((lane >> 4) * 8) + 2 * i + ((i >> 2) * 8);       // input channel (K)
  float f0 = w[(n * C_IN + c0) * 9 + tap];     // OIHW, tap = kh*3+kw
  float f1 = w[(n * C_IN + c0 + 1) * 9 + tap];
  wpk[idx] = (unsigned)f2bf(f0) | ((unsigned)f2bf(f1) << 16);
}

// ---------------------------------------------------------------------------
// Kernel 2: NCHW fp32 -> NHWC bf16. One block = one (n,h) row of 256 pixels;
// per-c reads are fully coalesced (lane varies w).
// ---------------------------------------------------------------------------
__global__ __launch_bounds__(256) void nchw_to_nhwc_bf16(const float* __restrict__ x,
                                                         unsigned short* __restrict__ dst) {
  int pix = blockIdx.x * 256 + threadIdx.x;              // n*65536 + h*256 + w
  const float* src = x + (size_t)(pix >> 16) * (C_IN * HSZ * WSZ) + (pix & 65535);
  unsigned int wds[16];
#pragma unroll
  for (int c = 0; c < C_IN; c += 2) {
    unsigned lo = f2bf(src[(size_t)c * (HSZ * WSZ)]);
    unsigned hi = f2bf(src[(size_t)(c + 1) * (HSZ * WSZ)]);
    wds[c >> 1] = lo | (hi << 16);
  }
  v4u* out = (v4u*)(dst + (size_t)pix * 32);
#pragma unroll
  for (int k = 0; k < 4; ++k) {
    v4u t = { wds[4 * k], wds[4 * k + 1], wds[4 * k + 2], wds[4 * k + 3] };
    out[k] = t;
  }
}

// ---------------------------------------------------------------------------
// Kernel 3: WMMA conv. One wave per (n,h) row; 16 tiles of 16 pixels along w.
// Per tile: 9 taps x 2 output-channel tiles = 18 x v_wmma_f32_16x16x32_bf16.
// Weights: TDM (tensor_load_to_lds) -> 18 KB LDS. Epilogue: per-wave LDS
// transpose so every global store writes complete 64B output lines.
// ---------------------------------------------------------------------------
struct SMem {
  v16bf wlds[9 * 2 * 32];        // 18 KB B tiles   (at LDS offset 0)
  float otile[8][O_CH][16];      // 16 KB per-wave output transpose staging
};

__global__ __launch_bounds__(256) void conv_wmma(const unsigned short* __restrict__ xb_bits,
                                                 const unsigned int* __restrict__ wpk,
                                                 const float* __restrict__ bias,
                                                 const float* __restrict__ scale_p,
                                                 float* __restrict__ out) {
  __shared__ SMem sm;
  int tid = threadIdx.x;

#if __has_builtin(__builtin_amdgcn_tensor_load_to_lds)
  // --- TDM: DMA the 18KB packed-weight region into LDS (wave 0 issues) -----
  if (tid < 32) {
    unsigned long long ga = (unsigned long long)wpk;
    v4u g0 = { 1u,                                   // count=1 (valid user D#)
               0u,                                   // lds_addr: sm.wlds @ LDS base
               (unsigned)ga,                         // global_addr[31:0]
               (unsigned)(ga >> 32) | (2u << 30) };  // global_addr[56:32] | type=2
    v8i g1 = { (int)(2u << 16),                      // wg_mask=0, data_size=4B
               (int)((WPK_DWORDS & 0xFFFFu) << 16),  // tensor_dim0[15:0] @bits63:48
               (int)(1u << 16),                      // tensor_dim1=1    @bits95:80
               (int)(WPK_DWORDS << 16),              // tile_dim0        @bits127:112
               1,                                    // tile_dim1=1      @bits143:128
               WPK_DWORDS,                           // tensor_dim0_stride low32
               0, 0 };
    v4i z4 = { 0, 0, 0, 0 };
    v8i z8 = { 0, 0, 0, 0, 0, 0, 0, 0 };
    __builtin_amdgcn_tensor_load_to_lds(g0, g1, z4, z4, z8, 0);  // 6-arg form
    __builtin_amdgcn_s_wait_tensorcnt(0);            // s_wait_tensorcnt 0x0
  }
#else
  {
    unsigned int* l = (unsigned int*)sm.wlds;
#pragma unroll
    for (int k = 0; k < 18; ++k)
      l[tid + 256 * k] = wpk[tid + 256 * k];
  }
#endif
  __syncthreads();

  int lane      = tid & 31;
  int waveInBlk = tid >> 5;
  int row = blockIdx.x * 8 + waveInBlk;              // 0..4095 => (n,h)
  int n   = row >> 8;
  int h   = row & 255;
  int m   = lane & 15;                               // A row (pixel) / D col (o)
  int hl  = lane >> 4;                               // K-half selector

  float scale = scale_p[0];
  float b0v   = bias[m];
  float b1v   = bias[m + 16];

  const __bf16* xb = (const __bf16*)xb_bits;
  float* mytile = &sm.otile[waveInBlk][0][0];        // [32][16] floats, this wave

  for (int wt = 0; wt < 16; ++wt) {
    v8f acc0 = (v8f)0.0f;
    v8f acc1 = (v8f)0.0f;

    if (wt < 15) {  // prefetch next tile's center row -> global_prefetch_b8
      const __bf16* pf = xb + ((size_t)((n << 8) + h) * 256 + (wt + 1) * 16 + m) * 32;
      __builtin_prefetch(pf, 0, 0);
    }

#pragma unroll
    for (int tap = 0; tap < 9; ++tap) {
      int dh = tap / 3 - 1, dw = tap % 3 - 1;
      int hh = h + dh;
      int ww = wt * 16 + m + dw;
      bool valid = ((unsigned)hh < (unsigned)HSZ) & ((unsigned)ww < (unsigned)WSZ);
      int pixoff = ((n << 8) + hh) * 256 + ww;       // NHWC pixel index
      long eoff  = (long)(valid ? pixoff : 0) * 32 + hl * 8;
      v4u lo = *(const v4u*)(xb + eoff);             // K [8*hl, 8*hl+8)
      v4u hi = *(const v4u*)(xb + eoff + 16);        // K [16+8*hl, ...+8)
      if (!valid) { lo = (v4u)0u; hi = (v4u)0u; }    // branchless zero-pad
      v8u   au = __builtin_shufflevector(lo, hi, 0, 1, 2, 3, 4, 5, 6, 7);
      v16bf a  = __builtin_bit_cast(v16bf, au);
      v16bf bb0 = sm.wlds[(tap * 2 + 0) * 32 + lane];  // ds_load_b128 x2
      v16bf bb1 = sm.wlds[(tap * 2 + 1) * 32 + lane];
      acc0 = __builtin_amdgcn_wmma_f32_16x16x32_bf16(false, a, false, bb0,
                                                     (short)0, acc0, false, false);
      acc1 = __builtin_amdgcn_wmma_f32_16x16x32_bf16(false, a, false, bb1,
                                                     (short)0, acc1, false, false);
    }

    // --- Epilogue: apply scale/bias, transpose through LDS, line-coalesced ---
    // D layout: lane = N (o = m / m+16), VGPR r -> M = r + 8*hl (pixel w-offset)
    {
      v4f s0a, s0b, s1a, s1b;
#pragma unroll
      for (int r = 0; r < 4; ++r) {
        s0a[r] = acc0[r] * scale + b0v;
        s0b[r] = acc0[r + 4] * scale + b0v;
        s1a[r] = acc1[r] * scale + b1v;
        s1b[r] = acc1[r + 4] * scale + b1v;
      }
      int wbase = 8 * hl;                            // local w offset 0 or 8
      *(v4f*)&mytile[m * 16 + wbase]            = s0a;   // ds_store_b128 x4
      *(v4f*)&mytile[m * 16 + wbase + 4]        = s0b;
      *(v4f*)&mytile[(m + 16) * 16 + wbase]     = s1a;
      *(v4f*)&mytile[(m + 16) * 16 + wbase + 4] = s1b;
    }
    // Wave-synchronous readback: 4 global_store_b128, each instruction covers
    // 8 output channels' complete 64B lines (lanes 0-3 = one o's 16 pixels).
#pragma unroll
    for (int k = 0; k < 4; ++k) {
      int o  = k * 8 + (lane >> 2);
      int wl = (lane & 3) * 4;
      v4f v = *(const v4f*)&mytile[o * 16 + wl];     // ds_load_b128
      *(v4f*)(out + ((size_t)(n * O_CH + o) << 16) + h * 256 + wt * 16 + wl) = v;
    }
  }
}

// ---------------------------------------------------------------------------
extern "C" void kernel_launch(void* const* d_in, const int* in_sizes, int n_in,
                              void* d_out, int out_size, void* d_ws, size_t ws_size,
                              hipStream_t stream) {
  const float* x     = (const float*)d_in[0];   // (16,32,256,256)
  const float* w     = (const float*)d_in[1];   // (32,32,3,3)
  const float* bias  = (const float*)d_in[2];   // (32,)
  const float* scale = (const float*)d_in[3];   // scalar
  float* out = (float*)d_out;

  unsigned int*   wpk = (unsigned int*)d_ws;                         // 18 KB packed B
  unsigned short* xb  = (unsigned short*)((char*)d_ws + XOFF_BYTES); // 64 MB NHWC bf16

  pack_weights     <<<18,   256, 0, stream>>>(w, wpk);
  nchw_to_nhwc_bf16<<<4096, 256, 0, stream>>>(x, xb);
  conv_wmma        <<<512,  256, 0, stream>>>(xb, wpk, bias, scale, out);
}
